// RiemannianCliffordConv2d_19902878449919
// MI455X (gfx1250) — matrix operand order repaired and checked by reading, not compiled, not run on hardware
//
#include <hip/hip_runtime.h>

typedef __attribute__((ext_vector_type(16))) __bf16 v16bf;
typedef __attribute__((ext_vector_type(8)))  float  v8f;
typedef __attribute__((ext_vector_type(4)))  unsigned int v4u;
typedef __attribute__((ext_vector_type(8)))  int v8i;
typedef __attribute__((ext_vector_type(4)))  int v4i;

#define Hh 128
#define Ww 128
#define HW 16384
#define Cc 128
#define Oo 256
#define Bb 16

__device__ __forceinline__ unsigned short f2bf(float f) {
  unsigned u = __float_as_uint(f);
  u += 0x7fffu + ((u >> 16) & 1u);          // round-to-nearest-even
  return (unsigned short)(u >> 16);
}

__device__ __forceinline__ float sigmoidf(float t) {
  return 1.0f / (1.0f + __expf(-t));
}

union FragU { uint4 q[2]; v16bf v; };

// A-fragment: 16 contiguous bf16 (one row's 16 K values)
__device__ __forceinline__ v16bf ld_fragA(const unsigned short* p) {
  FragU f;
  f.q[0] = *(const uint4*)(p);
  f.q[1] = *(const uint4*)(p + 8);
  return f.v;
}
// B-fragment: 16 contiguous bf16 (one column's 16 K values)
__device__ __forceinline__ v16bf ld_fragB(const unsigned short* p) {
  FragU f;
  f.q[0] = *(const uint4*)(p);
  f.q[1] = *(const uint4*)(p + 8);
  return f.v;
}

// ---------------------------------------------------------------------------
// Tensor Data Mover: 2D bf16 tile (tile_d0 elems/row, tile_d1 rows, global row
// stride in elements) -> LDS, rows packed contiguously.
// D# per CDNA5 ISA (data_size=1 -> 2 bytes, type=2).
// 6-arg builtin variant (clang-23 / therock headers).
// Issue from ONE wave; caller must s_wait_tensorcnt(0) + barrier.
// ---------------------------------------------------------------------------
__device__ __forceinline__ void tdm_load_2d_bf16(
    unsigned lds_addr, const unsigned short* gsrc,
    unsigned tile_d0, unsigned tile_d1, unsigned row_stride)
{
  const unsigned long long ga = (unsigned long long)(const void*)gsrc;
  v4u g0;
  g0[0] = 1u;                                              // count=1 (valid), user
  g0[1] = lds_addr;                                        // LDS byte address
  g0[2] = (unsigned)(ga & 0xffffffffu);                    // global_addr[31:0]
  g0[3] = (unsigned)((ga >> 32) & 0x1ffffffu) | (2u << 30); // addr[56:32] | type=2
  v8i g1;
  g1[0] = (int)(1u << 16);                                 // wg_mask=0, data_size=1 (2B)
  g1[1] = (int)((row_stride & 0xffffu) << 16);             // tensor_dim0[15:0]
  g1[2] = (int)((row_stride >> 16) | ((tile_d1 & 0xffffu) << 16)); // td0[31:16]|td1[15:0]
  g1[3] = (int)((tile_d1 >> 16) | (tile_d0 << 16));        // td1[31:16] | tile_dim0
  g1[4] = (int)(tile_d1 & 0xffffu);                        // tile_dim1 | tile_dim2=0
  g1[5] = (int)row_stride;                                 // tensor_dim0_stride[31:0]
  g1[6] = 0;                                               // stride[47:32] | dim1_stride lo
  g1[7] = 0;
  v4i g2 = {0, 0, 0, 0};                                   // 2D: dims 2/3 unused
  v4i g3 = {0, 0, 0, 0};
  v8i g4 = {0, 0, 0, 0, 0, 0, 0, 0};                       // 6-arg form: extra group
  __builtin_amdgcn_tensor_load_to_lds(g0, g1, g2, g3, g4, 0);
}

// ---------------------------------------------------------------------------
// Kernel 0: prep — gm table, bf16 weight conversions, conv_w transpose
// dst conv layout: wcbf[((o*3+ky)*3+kx)*128 + c]
// ---------------------------------------------------------------------------
__global__ __launch_bounds__(256) void prep_kernel(
    const float* __restrict__ conv_w, const float* __restrict__ metric_field,
    const float* __restrict__ visc_w, const float* __restrict__ anti_w,
    const float* __restrict__ metric_scale,
    float* __restrict__ gm, unsigned short* __restrict__ wvbf,
    unsigned short* __restrict__ wabf, unsigned short* __restrict__ wcbf)
{
  const int i = blockIdx.x * 256 + threadIdx.x;
  if (i < HW) gm[i] = __expf(metric_field[i] * metric_scale[0]);
  if (i < Cc * Cc) { wvbf[i] = f2bf(visc_w[i]); wabf[i] = f2bf(anti_w[i]); }
  if (i < Oo * Cc * 9) {
    int c = i & 127;
    int r = i >> 7;
    int kx = r % 3; r /= 3;
    int ky = r % 3; r /= 3;
    int o = r;
    wcbf[i] = f2bf(conv_w[((o * Cc + c) * 3 + ky) * 3 + kx]);
  }
}

// ---------------------------------------------------------------------------
// Kernel 1: per-(b,c) spatial mean
// ---------------------------------------------------------------------------
__global__ __launch_bounds__(256) void means_kernel(
    const float* __restrict__ x, float* __restrict__ means)
{
  __shared__ float red[256];
  const int bc = blockIdx.x;
  const float* p = x + (size_t)bc * HW;
  float s = 0.0f;
  for (int i = threadIdx.x; i < HW; i += 256) s += p[i];
  red[threadIdx.x] = s;
  __syncthreads();
  for (int st = 128; st > 0; st >>= 1) {
    if (threadIdx.x < st) red[threadIdx.x] += red[threadIdx.x + st];
    __syncthreads();
  }
  if (threadIdx.x == 0) means[bc] = red[0] * (1.0f / (float)HW);
}

// ---------------------------------------------------------------------------
// Kernel 2: gc[b,c] = 1 + 0.1*tanh(mean[b,:]·refiner_w[c,:] + refiner_b[c])
// ---------------------------------------------------------------------------
__global__ __launch_bounds__(128) void gc_kernel(
    const float* __restrict__ means, const float* __restrict__ rw,
    const float* __restrict__ rb, float* __restrict__ gc)
{
  __shared__ float m[Cc];
  const int b = blockIdx.x, c = threadIdx.x;
  m[c] = means[b * Cc + c];
  __syncthreads();
  float s = rb[c];
  const float* wr = rw + c * Cc;
  for (int k = 0; k < Cc; ++k) s += m[k] * wr[k];
  gc[b * Cc + c] = 1.0f + 0.1f * tanhf(s);
}

// ---------------------------------------------------------------------------
// Kernel 3: fused 1x1 GEMMs (bf16 WMMA) + laplacian + elementwise -> xf (bf16)
// Tile: one batch b, one row h, 64 pixels. M=C=128, K=C=128, N=64.
// Weight K-slices staged by TDM (tensor_load_to_lds), X tile by VALU convert.
// ---------------------------------------------------------------------------
__global__ __launch_bounds__(256) void stage_kernel(
    const float* __restrict__ x, const float* __restrict__ gm,
    const float* __restrict__ gc,
    const unsigned short* __restrict__ wvbf, const unsigned short* __restrict__ wabf,
    const float* __restrict__ visc_b, const float* __restrict__ anti_b,
    const float* __restrict__ viscosity_scale, const float* __restrict__ dispersion_scale,
    const float* __restrict__ diffusion_scale,
    unsigned short* __restrict__ xf)
{
  __shared__ unsigned short xs[64 * 136];     // [pixel][c], 16B-aligned rows
  __shared__ unsigned short wt[2][128][32];   // K-step tiles: 0=visc, 1=anti
  __shared__ float gcl[Cc], vbl[Cc], abl[Cc];

  const int t   = threadIdx.x;
  const int b   = blockIdx.y;
  const int h   = blockIdx.x >> 1;
  const int w0  = (blockIdx.x & 1) * 64;

  // stage X tile (f32 -> bf16) into LDS [pix][c]
  {
    const int c = t >> 1;
    const int p0 = (t & 1) * 32;
    const float* src = x + (((size_t)(b * Cc + c)) << 14) + h * Ww + w0 + p0;
    for (int i = 0; i < 32; ++i)
      xs[(p0 + i) * 136 + c] = f2bf(src[i]);
  }
  if (t < Cc) { gcl[t] = gc[b * Cc + t]; vbl[t] = visc_b[t]; abl[t] = anti_b[t]; }

  const int lane = t & 31, wave = t >> 5;
  const int lrow = lane & 15, half = lane >> 4;
  const unsigned wt0_lds = (unsigned)(unsigned long long)(void*)&wt[0][0][0];
  const unsigned wt1_lds = (unsigned)(unsigned long long)(void*)&wt[1][0][0];

  const v8f vzero = {0.f, 0.f, 0.f, 0.f, 0.f, 0.f, 0.f, 0.f};
  v8f accv[4], acca[4];
#pragma unroll
  for (int i = 0; i < 4; ++i) { accv[i] = vzero; acca[i] = vzero; }

  for (int ks = 0; ks < 4; ++ks) {
    const int k0 = ks * 32;
    __syncthreads();   // (ks=0) xs staged; (ks>0) previous frag reads done
    if (wave == 0) {   // TDM: two 128x32 bf16 tiles, row stride 128
      tdm_load_2d_bf16(wt0_lds, wvbf + k0, 32, 128, Cc);
      tdm_load_2d_bf16(wt1_lds, wabf + k0, 32, 128, Cc);
      __builtin_amdgcn_s_wait_tensorcnt(0);
    }
    __syncthreads();
    const v16bf av = ld_fragA(&wt[0][16 * wave + lrow][half * 16]);
    const v16bf aa = ld_fragA(&wt[1][16 * wave + lrow][half * 16]);
#pragma unroll
    for (int nb = 0; nb < 4; ++nb) {
      const v16bf bf = ld_fragB(&xs[(16 * nb + lrow) * 136 + k0 + half * 16]);
      accv[nb] = __builtin_amdgcn_wmma_f32_16x16x32_bf16(
          false, av, false, bf, (short)0, accv[nb], false, false);
      acca[nb] = __builtin_amdgcn_wmma_f32_16x16x32_bf16(
          false, aa, false, bf, (short)0, acca[nb], false, false);
    }
  }

  const float vds = viscosity_scale[0] * diffusion_scale[0];
  const float dsp = dispersion_scale[0];
#pragma unroll
  for (int nb = 0; nb < 4; ++nb) {
#pragma unroll
    for (int r = 0; r < 8; ++r) {
      const int c  = 16 * wave + half * 8 + r;
      const int wp = w0 + 16 * nb + lrow;
      const float gcc = gcl[c];
      const float* xp = x + (((size_t)(b * Cc + c)) << 14);
      const float xc  = xp[h * Ww + wp];
      const float g   = gm[h * Ww + wp] * gcc;
      const float xsc = xc * g;
      float lap = -4.0f * xsc;
      if (h > 0)      lap += xp[(h - 1) * Ww + wp] * gm[(h - 1) * Ww + wp] * gcc;
      if (h < Hh - 1) lap += xp[(h + 1) * Ww + wp] * gm[(h + 1) * Ww + wp] * gcc;
      if (wp > 0)     lap += xp[h * Ww + wp - 1]  * gm[h * Ww + wp - 1]  * gcc;
      if (wp < Ww - 1)lap += xp[h * Ww + wp + 1]  * gm[h * Ww + wp + 1]  * gcc;
      const float inv = 1.0f / (g + 1e-6f);
      const float v   = sigmoidf(accv[nb][r] + vbl[c]) * inv;
      const float dif = lap * v * vds;
      const float a   = sigmoidf(acca[nb][r] + abl[c]);
      const float ev  = xsc + dif - 0.5f * a * dif - lap * dsp;
      xf[(((size_t)(b * Cc + c)) << 14) + h * Ww + wp] = f2bf(ev * inv);
    }
  }
}

// ---------------------------------------------------------------------------
// Kernel 4: 3x3 conv as 9 shifted K=128 bf16 WMMA GEMMs, fp32 accum.
// Tile: (b, h, o-block of 128). M=128, N=128 (full row), K=9*128.
// LDS: input row (+halo) [w+1][c]; weight tap K-slice via TDM.
// ---------------------------------------------------------------------------
__global__ __launch_bounds__(256) void conv3_kernel(
    const unsigned short* __restrict__ xf, const unsigned short* __restrict__ wcbf,
    float* __restrict__ out)
{
  __shared__ unsigned short xsl[130 * 136];   // [w+1][c]
  __shared__ unsigned short wa[128 * 32];     // [m][k-slice]

  const int t  = threadIdx.x;
  const int h  = blockIdx.x;
  const int b  = blockIdx.y;
  const int o0 = blockIdx.z * 128;

  const int lane = t & 31, wave = t >> 5;
  const int lrow = lane & 15, half = lane >> 4;
  const unsigned wa_lds = (unsigned)(unsigned long long)(void*)&wa[0];

  const v8f vzero = {0.f, 0.f, 0.f, 0.f, 0.f, 0.f, 0.f, 0.f};
  v8f acc[8];
#pragma unroll
  for (int i = 0; i < 8; ++i) acc[i] = vzero;

  for (int ky = 0; ky < 3; ++ky) {
    const int row = h + ky - 1;
    __syncthreads();   // previous xsl readers done
    {
      const int c = t >> 1, hh = t & 1;
      if (row >= 0 && row < Hh) {
        const unsigned short* src = xf + (((size_t)(b * Cc + c)) << 14) + row * Ww;
        for (int i = 0; i < 65; ++i) {
          const int w = hh * 65 + i - 1;
          xsl[(w + 1) * 136 + c] = (w >= 0 && w < Ww) ? src[w] : (unsigned short)0;
        }
      } else {
        for (int i = 0; i < 65; ++i) {
          const int w = hh * 65 + i - 1;
          xsl[(w + 1) * 136 + c] = 0;
        }
      }
    }
    for (int kx = 0; kx < 3; ++kx) {
      for (int ks = 0; ks < 4; ++ks) {
        __syncthreads();
        if (wave == 0) {   // TDM: 128x32 bf16 weight tile, row stride 1152
          const unsigned short* src =
              wcbf + (((size_t)(o0 * 3 + ky) * 3 + kx) * 128) + ks * 32;
          tdm_load_2d_bf16(wa_lds, src, 32, 128, 9 * Cc);
          __builtin_amdgcn_s_wait_tensorcnt(0);
        }
        __syncthreads();
        const v16bf af = ld_fragA(&wa[(16 * wave + lrow) * 32 + half * 16]);
#pragma unroll
        for (int nb = 0; nb < 8; ++nb) {
          const v16bf bf =
              ld_fragB(&xsl[(16 * nb + lrow + kx) * 136 + ks * 32 + half * 16]);
          acc[nb] = __builtin_amdgcn_wmma_f32_16x16x32_bf16(
              false, af, false, bf, (short)0, acc[nb], false, false);
        }
      }
    }
  }

#pragma unroll
  for (int nb = 0; nb < 8; ++nb) {
#pragma unroll
    for (int r = 0; r < 8; ++r) {
      const int o  = o0 + 16 * wave + half * 8 + r;
      const int wp = 16 * nb + lrow;
      out[(((size_t)(b * Oo + o)) << 14) + h * Ww + wp] = acc[nb][r];
    }
  }
}

// ---------------------------------------------------------------------------
// Launch
// ---------------------------------------------------------------------------
extern "C" void kernel_launch(void* const* d_in, const int* in_sizes, int n_in,
                              void* d_out, int out_size, void* d_ws, size_t ws_size,
                              hipStream_t stream) {
  (void)in_sizes; (void)n_in; (void)out_size; (void)ws_size;
  const float* x   = (const float*)d_in[0];
  const float* cw  = (const float*)d_in[1];
  const float* mf  = (const float*)d_in[2];
  const float* rw  = (const float*)d_in[3];
  const float* rb  = (const float*)d_in[4];
  const float* vw  = (const float*)d_in[5];
  const float* vb  = (const float*)d_in[6];
  const float* aw  = (const float*)d_in[7];
  const float* ab  = (const float*)d_in[8];
  const float* ms  = (const float*)d_in[9];
  const float* vs  = (const float*)d_in[10];
  const float* ds  = (const float*)d_in[11];
  const float* dfs = (const float*)d_in[12];
  float* out = (float*)d_out;

  char* ws = (char*)d_ws;
  float* means         = (float*)(ws + 0);            //  8 KB
  float* gc            = (float*)(ws + 8192);         //  8 KB
  float* gm            = (float*)(ws + 16384);        // 64 KB
  unsigned short* wvbf = (unsigned short*)(ws + 81920);   // 32 KB
  unsigned short* wabf = (unsigned short*)(ws + 114688);  // 32 KB
  unsigned short* wcbf = (unsigned short*)(ws + 147456);  // 576 KB
  unsigned short* xf   = (unsigned short*)(ws + 737280);  // 64 MB

  prep_kernel<<<1152, 256, 0, stream>>>(cw, mf, vw, aw, ms, gm, wvbf, wabf, wcbf);
  means_kernel<<<Bb * Cc, 256, 0, stream>>>(x, means);
  gc_kernel<<<Bb, Cc, 0, stream>>>(means, rw, rb, gc);
  stage_kernel<<<dim3(256, Bb), 256, 0, stream>>>(
      x, gm, gc, wvbf, wabf, vb, ab, vs, ds, dfs, xf);
  conv3_kernel<<<dim3(Hh, Bb, 2), 256, 0, stream>>>(xf, wcbf, out);
}
